// GaussianModel_5798205850208
// MI455X (gfx1250) — matrix operand order repaired and checked by reading, not compiled, not run on hardware
//
#include <hip/hip_runtime.h>
#include <math.h>

typedef __bf16 bf16;
typedef __attribute__((ext_vector_type(16))) __bf16 v16bf;
typedef __attribute__((ext_vector_type(8)))  __bf16 v8bf;
typedef __attribute__((ext_vector_type(8)))  float  v8f;
typedef __attribute__((address_space(3))) bf16 lds_bf16_t;

#define BM 128
#define BN 128
#define BK 32

__device__ __forceinline__ unsigned lds_offset(const bf16* p)
{
    // generic -> LDS addrspace cast strips the aperture; low 32 bits are the
    // wave-relative LDS byte offset expected by GLOBAL_LOAD_ASYNC_TO_LDS.
    return (unsigned)(size_t)(lds_bf16_t*)p;
}

// ---------------------------------------------------------------------------
// Generic bf16 WMMA GEMM: C[M,N] = A[M,K] @ B  (+bias, optional relu)
//   bT==0: B is [K,N] row-major (NN)
//   bT==1: B is [N,K] row-major, effective op is A @ B^T (Gram)
// Writes float C (if non-null) and/or bf16 Cb (if non-null).
// Block = 256 threads = 8 waves; block tile 128x128; wave tile 32x64.
// A-tile (and NT B-tile) staged with GLOBAL_LOAD_ASYNC_TO_LDS_B128 (ASYNCcnt).
// ---------------------------------------------------------------------------
__global__ __launch_bounds__(256)
void wmma_gemm(const bf16* __restrict__ A, const bf16* __restrict__ B,
               float* __restrict__ C, bf16* __restrict__ Cb,
               const float* __restrict__ bias,
               int M, int N, int K, int relu, int bT)
{
    __shared__ bf16 As[BM * BK];   // row-major [m][k]
    __shared__ bf16 Bs[BN * BK];   // N-major   [n][k]

    const int tid  = threadIdx.x;
    const int lane = tid & 31;
    const int wave = tid >> 5;            // 0..7
    const int wMb  = (wave & 3) * 32;     // wave row offset in block tile
    const int wNb  = (wave >> 2) * 64;    // wave col offset in block tile
    const int mBlk = blockIdx.y * BM;
    const int nBlk = blockIdx.x * BN;

    const int lr = lane & 15;             // lane row/col within half
    const int lh = lane >> 4;             // lane half select

    const unsigned asBase = lds_offset(As);
    const unsigned bsBase = lds_offset(Bs);

    v8f zero{};
    v8f acc[2][4];
#pragma unroll
    for (int i = 0; i < 2; ++i)
#pragma unroll
        for (int j = 0; j < 4; ++j) acc[i][j] = zero;

    for (int k0 = 0; k0 < K; k0 += BK) {
        // ---- fill A tile: 128 rows x 32 k, async 16B per thread x2 ----
        {
            const int row = tid >> 2;            // 0..63
            const int seg = (tid & 3) * 8;       // 0,8,16,24
#pragma unroll
            for (int p = 0; p < 2; ++p) {
                const int r = row + p * 64;
                const bf16* src = A + (size_t)(mBlk + r) * K + k0 + seg;
                const unsigned dst = asBase + (unsigned)((r * BK + seg) * sizeof(bf16));
                asm volatile("global_load_async_to_lds_b128 %0, %1, off"
                             :: "v"(dst), "v"(src) : "memory");
            }
        }
        // ---- fill B tile into N-major layout ----
        if (bT) {
            const int row = tid >> 2;
            const int seg = (tid & 3) * 8;
#pragma unroll
            for (int p = 0; p < 2; ++p) {
                const int r = row + p * 64;
                const bf16* src = B + (size_t)(nBlk + r) * K + k0 + seg;
                const unsigned dst = bsBase + (unsigned)((r * BK + seg) * sizeof(bf16));
                asm volatile("global_load_async_to_lds_b128 %0, %1, off"
                             :: "v"(dst), "v"(src) : "memory");
            }
        } else {
            const int kk = tid >> 5;             // 0..7
            const int n4 = (tid & 31) * 4;       // 0..124
#pragma unroll
            for (int p = 0; p < 4; ++p) {
                const int k = kk + p * 8;
                const bf16* src = B + (size_t)(k0 + k) * N + nBlk + n4;
                bf16 v0 = src[0], v1 = src[1], v2 = src[2], v3 = src[3];
                Bs[(n4 + 0) * BK + k] = v0;
                Bs[(n4 + 1) * BK + k] = v1;
                Bs[(n4 + 2) * BK + k] = v2;
                Bs[(n4 + 3) * BK + k] = v3;
            }
        }
        asm volatile("s_wait_asynccnt 0" ::: "memory");
        __syncthreads();

        // ---- load fragments per ISA 16-bit layouts ----
        // A 16x32: lanes 0-15 row M=lr K{0..7,16..23}; lanes 16-31 K{8..15,24..31}
        v16bf afrag[2];
#pragma unroll
        for (int i = 0; i < 2; ++i) {
            const int r = wMb + i * 16 + lr;
            v8bf lo = *(const v8bf*)&As[r * BK + lh * 8];
            v8bf hi = *(const v8bf*)&As[r * BK + lh * 8 + 16];
#pragma unroll
            for (int e = 0; e < 8; ++e) { afrag[i][e] = lo[e]; afrag[i][e + 8] = hi[e]; }
        }
        // B 32x16: lanes 0-15 col N=lr K{0..7,16..23}; lanes 16-31 K{8..15,24..31}
        v16bf bfrag[4];
#pragma unroll
        for (int j = 0; j < 4; ++j) {
            const int n = wNb + j * 16 + lr;
            v8bf lo = *(const v8bf*)&Bs[n * BK + lh * 8];
            v8bf hi = *(const v8bf*)&Bs[n * BK + lh * 8 + 16];
#pragma unroll
            for (int e = 0; e < 8; ++e) { bfrag[j][e] = lo[e]; bfrag[j][e + 8] = hi[e]; }
        }
#pragma unroll
        for (int i = 0; i < 2; ++i)
#pragma unroll
            for (int j = 0; j < 4; ++j)
                acc[i][j] = __builtin_amdgcn_wmma_f32_16x16x32_bf16(
                    false, afrag[i], false, bfrag[j], (short)0, acc[i][j], false, false);
        __syncthreads();
    }

    // ---- epilogue: D 16x16 f32 layout: VGPR r -> M = r + lh*8, N = lr ----
#pragma unroll
    for (int i = 0; i < 2; ++i) {
        const int m0 = mBlk + wMb + i * 16 + lh * 8;
#pragma unroll
        for (int j = 0; j < 4; ++j) {
            const int col = nBlk + wNb + j * 16 + lr;
            const float bv = bias ? bias[col] : 0.0f;
#pragma unroll
            for (int r = 0; r < 8; ++r) {
                float v = acc[i][j][r] + bv;
                if (relu) v = fmaxf(v, 0.0f);
                const size_t off = (size_t)(m0 + r) * N + col;
                if (C)  C[off]  = v;
                if (Cb) Cb[off] = (bf16)v;
            }
        }
    }
}

// ---------------------------------------------------------------------------
__global__ void f2bf_kernel(const float* __restrict__ src, bf16* __restrict__ dst, size_t n)
{
    size_t i = (size_t)blockIdx.x * blockDim.x + threadIdx.x;
    if (i < n) dst[i] = (bf16)src[i];
}

__device__ __forceinline__ float block_reduce_sum(float v, float* red)
{
    const int tid = threadIdx.x;
    red[tid] = v; __syncthreads();
    for (int st = 128; st > 0; st >>= 1) {
        if (tid < st) red[tid] += red[tid + st];
        __syncthreads();
    }
    float r = red[0]; __syncthreads();
    return r;
}

// Per-row stats over x_mean/x_std [N,F]:
//  m = l2norm(x_mean) -> bf16 ; sq = sum(m*m)
//  c = l2norm(exp(x_std)) ; csum = sum(c) ; cs = sqrt(c) -> bf16
__global__ __launch_bounds__(256)
void row_stats(const float* __restrict__ xm, const float* __restrict__ xs,
               bf16* __restrict__ m_bf, bf16* __restrict__ cs_bf,
               float* __restrict__ sq, float* __restrict__ csum, int F)
{
    __shared__ float red[256];
    const int row = blockIdx.x;
    const int tid = threadIdx.x;
    const float* xr = xm + (size_t)row * F;
    const float* sr = xs + (size_t)row * F;

    float s = 0.f;
    for (int f = tid; f < F; f += 256) { float v = xr[f]; s += v * v; }
    float nm = fmaxf(sqrtf(block_reduce_sum(s, red)), 1e-12f);

    float sqa = 0.f;
    for (int f = tid; f < F; f += 256) {
        float v = xr[f] / nm;
        m_bf[(size_t)row * F + f] = (bf16)v;
        sqa += v * v;
    }
    float sqr = block_reduce_sum(sqa, red);
    if (tid == 0) sq[row] = sqr;

    float s2 = 0.f;
    for (int f = tid; f < F; f += 256) { float c = expf(sr[f]); s2 += c * c; }
    float nc = fmaxf(sqrtf(block_reduce_sum(s2, red)), 1e-12f);

    float ca = 0.f;
    for (int f = tid; f < F; f += 256) {
        float c = expf(sr[f]) / nc;
        cs_bf[(size_t)row * F + f] = (bf16)sqrtf(c);
        ca += c;
    }
    float cs = block_reduce_sum(ca, red);
    if (tid == 0) csum[row] = cs;
}

// wsraw = exp(-(max(sq_i+sq_j-2*G1,0) + csum_i + csum_j - 2*G2)), in-place into G1
__global__ __launch_bounds__(256)
void ws_elem(float* __restrict__ G1, const float* __restrict__ G2,
             const float* __restrict__ sq, const float* __restrict__ csum, int Nn)
{
    size_t idx = (size_t)blockIdx.x * blockDim.x + threadIdx.x;
    if (idx >= (size_t)Nn * Nn) return;
    int i = (int)(idx / Nn), j = (int)(idx % Nn);
    float d2  = fmaxf(sq[i] + sq[j] - 2.f * G1[idx], 0.f);
    float res = d2 + csum[i] + csum[j] - 2.f * G2[idx];
    G1[idx] = expf(-res);
}

__global__ __launch_bounds__(256)
void row_norm(const float* __restrict__ W, float* __restrict__ nrm, int Nn)
{
    __shared__ float red[256];
    const int row = blockIdx.x;
    float s = 0.f;
    for (int f = threadIdx.x; f < Nn; f += 256) { float v = W[(size_t)row * Nn + f]; s += v * v; }
    float t = block_reduce_sum(s, red);
    if (threadIdx.x == 0) nrm[row] = fmaxf(sqrtf(t), 1e-12f);
}

// Build sparsified dense adjacency A (incl. conditional self-loops) into Aout
__global__ __launch_bounds__(256)
void term_elem(const float* __restrict__ Wn, const float* __restrict__ nrm,
               const float* __restrict__ new_edge, const float* __restrict__ epsm,
               const float* __restrict__ beta, const float* __restrict__ delta,
               float* __restrict__ Aout, int Nn)
{
    size_t idx = (size_t)blockIdx.x * blockDim.x + threadIdx.x;
    if (idx >= (size_t)Nn * Nn) return;
    int i = (int)(idx / Nn), j = (int)(idx % Nn);
    const float b = beta[0], d = delta[0];
    float w = Wn[idx] / nrm[i];
    float t = (1.f - b) * w + b * new_edge[idx];
    t = fminf(fmaxf(t, 1e-6f), 1.f - 1e-6f);
    t = logf(t / (1.f - t));
    float e = fminf(fmaxf(epsm[idx], 1e-6f), 1.f - 1e-6f);
    t += logf(e / (1.f - e));
    t = 1.f / (1.f + expf(-t));
    t = (t > d) ? t : 0.f;
    if (i == j && t <= 0.f) t = 1.f;   // add_remaining_self_loops(fill=1)
    Aout[idx] = t;
}

// dis[j] = deg_j^{-1/2}, deg = column sums (coalesced: block of 256 columns)
__global__ __launch_bounds__(256)
void col_deg(const float* __restrict__ Amat, float* __restrict__ dis, int Nn)
{
    const int j = blockIdx.x * 256 + threadIdx.x;
    float s = 0.f;
    for (int i = 0; i < Nn; ++i) s += Amat[(size_t)i * Nn + j];
    dis[j] = (s > 0.f) ? rsqrtf(s) : 0.f;
}

// AnT[i,j] = dis[i] * A[j,i] * dis[j]  (LDS-tiled transpose, bf16 output)
__global__ void build_ant(const float* __restrict__ Amat, const float* __restrict__ dis,
                          bf16* __restrict__ AnT, int Nn)
{
    __shared__ float tile[32][33];
    const int ib = blockIdx.x * 32;   // output row (i) base
    const int jb = blockIdx.y * 32;   // output col (j) base
    const int tx = threadIdx.x;       // 0..31
    const int ty = threadIdx.y;       // 0..7
#pragma unroll
    for (int p = 0; p < 4; ++p) {
        int r = ty + p * 8;
        tile[r][tx] = Amat[(size_t)(jb + r) * Nn + ib + tx];   // A[j][i]
    }
    __syncthreads();
#pragma unroll
    for (int p = 0; p < 4; ++p) {
        int ii = ty + p * 8;
        float v = dis[ib + ii] * tile[tx][ii] * dis[jb + tx];
        AnT[(size_t)(ib + ii) * Nn + jb + tx] = (bf16)v;
    }
}

// ---------------------------------------------------------------------------
extern "C" void kernel_launch(void* const* d_in, const int* in_sizes, int n_in,
                              void* d_out, int out_size, void* d_ws, size_t ws_size,
                              hipStream_t stream)
{
    const int Nn = 4096, F = 512, H = 256, H2 = 512;

    const float* x        = (const float*)d_in[0];
    const float* new_edge = (const float*)d_in[1];
    const float* beta     = (const float*)d_in[2];
    const float* delta    = (const float*)d_in[3];
    const float* epsm     = (const float*)d_in[4];
    const float* Wm  = (const float*)d_in[5];  const float* bm  = (const float*)d_in[6];
    const float* Ws  = (const float*)d_in[7];  const float* bs  = (const float*)d_in[8];
    const float* mW0 = (const float*)d_in[9];  const float* mb0 = (const float*)d_in[10];
    const float* mW1 = (const float*)d_in[11]; const float* mb1 = (const float*)d_in[12];
    const float* sW0 = (const float*)d_in[13]; const float* sb0 = (const float*)d_in[14];
    const float* sW1 = (const float*)d_in[15]; const float* sb1 = (const float*)d_in[16];
    float* out = (float*)d_out;

    char* wsp = (char*)d_ws;
    size_t off = 0;
    auto alloc = [&](size_t bytes) -> void* {
        void* p = wsp + off;
        off = (off + bytes + 255) & ~(size_t)255;
        return p;
    };

    bf16* xbf      = (bf16*)alloc((size_t)Nn * F * 2);
    bf16* Wmbf     = (bf16*)alloc((size_t)F * F * 2);
    bf16* Wsbf     = (bf16*)alloc((size_t)F * F * 2);
    bf16* mW0bf    = (bf16*)alloc((size_t)F * H2 * 2);
    bf16* mW1bf    = (bf16*)alloc((size_t)H2 * H * 2);
    bf16* sW0bf    = (bf16*)alloc((size_t)F * H2 * 2);
    bf16* sW1bf    = (bf16*)alloc((size_t)H2 * H * 2);
    float* xmean   = (float*)alloc((size_t)Nn * F * 4);
    float* xstd    = (float*)alloc((size_t)Nn * F * 4);
    bf16* xmean_bf = (bf16*)alloc((size_t)Nn * F * 2);
    bf16* xstd_bf  = (bf16*)alloc((size_t)Nn * F * 2);
    bf16* m_bf     = (bf16*)alloc((size_t)Nn * F * 2);
    bf16* cs_bf    = (bf16*)alloc((size_t)Nn * F * 2);
    float* sq      = (float*)alloc((size_t)Nn * 4);
    float* csum    = (float*)alloc((size_t)Nn * 4);
    float* nrm     = (float*)alloc((size_t)Nn * 4);
    float* dis     = (float*)alloc((size_t)Nn * 4);
    float* G1      = (float*)alloc((size_t)Nn * Nn * 4);   // Gram(m) -> wsraw
    float* G2      = (float*)alloc((size_t)Nn * Nn * 4);   // Gram(cs) -> adjacency A
    bf16* AnT      = (bf16*)alloc((size_t)Nn * Nn * 2);
    bf16* t0       = (bf16*)alloc((size_t)Nn * H2 * 2);
    bf16* hb       = (bf16*)alloc((size_t)Nn * H2 * 2);
    bf16* t1       = (bf16*)alloc((size_t)Nn * H * 2);

    auto cvt = [&](const float* s, bf16* d, size_t n) {
        f2bf_kernel<<<(unsigned)((n + 255) / 256), 256, 0, stream>>>(s, d, n);
    };
    auto gemm = [&](const bf16* A, const bf16* B, float* C, bf16* Cb,
                    const float* bias, int M, int N, int K, int relu, int bT) {
        dim3 g(N / BN, M / BM);
        wmma_gemm<<<g, 256, 0, stream>>>(A, B, C, Cb, bias, M, N, K, relu, bT);
    };

    // 0) bf16 copies of inputs/weights
    cvt(x,   xbf,   (size_t)Nn * F);
    cvt(Wm,  Wmbf,  (size_t)F * F);
    cvt(Ws,  Wsbf,  (size_t)F * F);
    cvt(mW0, mW0bf, (size_t)F * H2);
    cvt(mW1, mW1bf, (size_t)H2 * H);
    cvt(sW0, sW0bf, (size_t)F * H2);
    cvt(sW1, sW1bf, (size_t)H2 * H);

    // 1) x_mean = x@Wm+bm ; x_std = x@Ws+bs  (fp32 + bf16 outputs)
    gemm(xbf, Wmbf, xmean, xmean_bf, bm, Nn, F, F, 0, 0);
    gemm(xbf, Wsbf, xstd,  xstd_bf,  bs, Nn, F, F, 0, 0);

    // 2) row statistics
    row_stats<<<Nn, 256, 0, stream>>>(xmean, xstd, m_bf, cs_bf, sq, csum, F);

    // 3) Gram matrices (A @ A^T) via WMMA-NT
    gemm(m_bf,  m_bf,  G1, nullptr, nullptr, Nn, Nn, F, 0, 1);
    gemm(cs_bf, cs_bf, G2, nullptr, nullptr, Nn, Nn, F, 0, 1);

    // 4) WS_v22 elementwise + row normalization
    const unsigned nnBlocks = (unsigned)(((size_t)Nn * Nn + 255) / 256);
    ws_elem<<<nnBlocks, 256, 0, stream>>>(G1, G2, sq, csum, Nn);
    row_norm<<<Nn, 256, 0, stream>>>(G1, nrm, Nn);

    // 5) sparsified adjacency into G2
    term_elem<<<nnBlocks, 256, 0, stream>>>(G1, nrm, new_edge, epsm, beta, delta, G2, Nn);

    // 6) GCN normalization: column degrees, then AnT = D^-1/2 A D^-1/2 transposed (bf16)
    col_deg<<<Nn / 256, 256, 0, stream>>>(G2, dis, Nn);
    build_ant<<<dim3(Nn / 32, Nn / 32), dim3(32, 8), 0, stream>>>(G2, dis, AnT, Nn);

    // 7) encoder(mean): relu(AnT@(x_mean@mW0)+mb0) -> relu(AnT@(h@mW1)+mb1)
    gemm(xmean_bf, mW0bf, nullptr, t0, nullptr, Nn, H2, F, 0, 0);
    gemm(AnT, t0, nullptr, hb, mb0, Nn, H2, Nn, 1, 0);
    gemm(hb, mW1bf, nullptr, t1, nullptr, Nn, H, H2, 0, 0);
    gemm(AnT, t1, out, nullptr, mb1, Nn, H, Nn, 1, 0);

    // 8) encoder(std)
    gemm(xstd_bf, sW0bf, nullptr, t0, nullptr, Nn, H2, F, 0, 0);
    gemm(AnT, t0, nullptr, hb, sb0, Nn, H2, Nn, 1, 0);
    gemm(hb, sW1bf, nullptr, t1, nullptr, Nn, H, H2, 0, 0);
    gemm(AnT, t1, out + (size_t)Nn * H, nullptr, sb1, Nn, H, Nn, 1, 0);
}
